// SubtasksRecurrence_64836826300621
// MI455X (gfx1250) — matrix operand order repaired and checked by reading, not compiled, not run on hardware
//
#include <hip/hip_runtime.h>
#include <hip/hip_bf16.h>
#include <math.h>

typedef __attribute__((ext_vector_type(16))) _Float16 v16h;
typedef __attribute__((ext_vector_type(8)))  float    v8f;

#define T_DIM     128
#define N_DIM     128
#define HID       64
#define CONV_OUT  4096
#define N_SUB     5
#define AA        5
#define BB        3
#define CC        6
#define SUB_SIZE  14
#define N_G       90
#define STATE     99
#define D_IN      4111     // CONV_OUT + 3*N_SUB
#define WF_STRIDE 4125     // CONV_OUT + 2*SUB_SIZE + 1
#define WB_STRIDE 4110     // CONV_OUT + SUB_SIZE
#define S_W       80       // padded GEMM output width (64 s_x + 2 beta_x + pad)
#define NT_TILES  5
#define KK_TILES  128      // 4096 / 32
#define WPACK_ELEMS (NT_TILES * KK_TILES * 32 * 16)  // 327680 halves

// ---------------------------------------------------------------------------
// Phase A1: pack [Wf[:, :4096]; Wb[:, :4096]; zero-pad] (80 x 4096) into f16
// B-matrix lane layout for v_wmma_f32_16x16x32_f16:
//   lane l holds column N = l&15; halves e=0..15 cover K = kk*32 + (l>=16?16:0) + e
// Flat index: ((nt*128 + kk)*32 + l)*16 + e   -> 32B aligned per-lane chunks.
// ---------------------------------------------------------------------------
__global__ void __launch_bounds__(256) pack_weights(const float* __restrict__ Wf,
                                                    const float* __restrict__ Wb,
                                                    _Float16* __restrict__ Wpack) {
  int idx = blockIdx.x * blockDim.x + threadIdx.x;
  if (idx >= WPACK_ELEMS) return;
  int e  = idx & 15;
  int l  = (idx >> 4) & 31;
  int kk = (idx >> 9) & 127;
  int nt = idx >> 16;
  int r = nt * 16 + (l & 15);                      // output column (weight row)
  int K = kk * 32 + ((l >> 4) ? 16 : 0) + e;       // reduction index
  float v = 0.0f;
  if (r < HID)          v = Wf[(size_t)r * WF_STRIDE + K];
  else if (r < HID + 2) v = Wb[(size_t)(r - HID) * WB_STRIDE + K];
  Wpack[idx] = (_Float16)v;
}

// ---------------------------------------------------------------------------
// Phase A2: new-episode flag = any(hx != 0)
// ---------------------------------------------------------------------------
__global__ void __launch_bounds__(256) check_hx(const float* __restrict__ hx,
                                                int* __restrict__ flag) {
  int i = blockIdx.x * blockDim.x + threadIdx.x;
  if (i < N_DIM * STATE && hx[i] != 0.0f) atomicOr(flag, 1);
}

// ---------------------------------------------------------------------------
// Phase B: S[row, 0:80] = obs[row, 0:4096] @ Wpacked^T  via v_wmma_f32_16x16x32_f16
// One wave per 16-row M-tile, all 5 N-tiles per wave (A reused 5x per K step).
// A-tile lane layout: lane l holds row M = l&15; halves 0..15 = K kb..kb+15,
// kb = kk*32 + (l>=16 ? 16 : 0).
// ---------------------------------------------------------------------------
__global__ void __launch_bounds__(256) gemm_sx(const float* __restrict__ input,
                                               const _Float16* __restrict__ Wpack,
                                               float* __restrict__ S) {
  const int tid   = threadIdx.x;
  const int lane  = tid & 31;
  const int wave  = tid >> 5;
  const int mtile = blockIdx.x * 8 + wave;
  const int row   = mtile * 16 + (lane & 15);
  const float* __restrict__ arow = input + (size_t)row * D_IN;   // obs row (cols 0..4095)
  const int khalf = lane >> 4;

  v8f c[NT_TILES] = {};
  for (int kk = 0; kk < KK_TILES; ++kk) {
    const int kb = kk * 32 + khalf * 16;
    v16h a;
#pragma unroll
    for (int i = 0; i < 8; ++i) a[i]     = (_Float16)arow[kb + i];
#pragma unroll
    for (int i = 0; i < 8; ++i) a[i + 8] = (_Float16)arow[kb + 8 + i];
    const size_t boff = ((size_t)kk * 32 + lane) * 16;
#pragma unroll
    for (int nt = 0; nt < NT_TILES; ++nt) {
      v16h b = *(const v16h*)(Wpack + (size_t)nt * 65536 + boff);
      c[nt] = __builtin_amdgcn_wmma_f32_16x16x32_f16(
          /*neg_a=*/false, a, /*neg_b=*/false, b,
          /*c_mod=*/(short)0, c[nt], /*reuse_a=*/false, /*reuse_b=*/false);
    }
  }
  // C/D layout: VGPR v -> lanes 0-15: M=v, lanes 16-31: M=v+8; N = lane&15
  const int col  = lane & 15;
  const int mofs = (lane >> 4) ? 8 : 0;
#pragma unroll
  for (int nt = 0; nt < NT_TILES; ++nt)
#pragma unroll
    for (int v = 0; v < 8; ++v)
      S[(size_t)(mtile * 16 + mofs + v) * S_W + nt * 16 + col] = c[nt][v];
}

// ---------------------------------------------------------------------------
// Phase C: the recurrence. One 128-thread block per batch row.
// ---------------------------------------------------------------------------
__device__ __forceinline__ float sigm(float x) { return 1.0f / (1.0f + expf(-x)); }

__device__ __forceinline__ float hash_u01(unsigned t, unsigned n, unsigned j, unsigned salt) {
  unsigned x = t * 0x9E3779B1u ^ n * 0x85EBCA77u ^ j * 0xC2B2AE3Du ^ (salt + 1u) * 0x27D4EB2Fu;
  x ^= x >> 17; x *= 0xED5AD4BBu; x ^= x >> 11; x *= 0xAC4C1B51u;
  x ^= x >> 15; x *= 0x31848BABu; x ^= x >> 14;
  return ((float)(x >> 8) + 0.5f) * (1.0f / 16777216.0f);   // (0,1)
}
__device__ __forceinline__ float gumbel(unsigned t, unsigned n, unsigned j, unsigned salt) {
  return -logf(-logf(hash_u01(t, n, j, salt)));
}

__device__ __forceinline__ float blk_sum(float v, float* red, int tid) {
  red[tid] = v; __syncthreads();
#pragma unroll
  for (int off = 64; off > 0; off >>= 1) {
    if (tid < off) red[tid] += red[tid + off];
    __syncthreads();
  }
  float r = red[0]; __syncthreads();
  return r;
}
__device__ __forceinline__ float blk_max(float v, float* red, int tid) {
  red[tid] = v; __syncthreads();
#pragma unroll
  for (int off = 64; off > 0; off >>= 1) {
    if (tid < off) red[tid] = fmaxf(red[tid], red[tid + off]);
    __syncthreads();
  }
  float r = red[0]; __syncthreads();
  return r;
}
__device__ __forceinline__ int blk_argmax(float v, float* red, int* redi, int tid) {
  red[tid] = v; redi[tid] = tid; __syncthreads();
#pragma unroll
  for (int off = 64; off > 0; off >>= 1) {
    if (tid < off && red[tid + off] > red[tid]) {
      red[tid] = red[tid + off]; redi[tid] = redi[tid + off];
    }
    __syncthreads();
  }
  int r = redi[0]; __syncthreads();
  return r;
}

__global__ void __launch_bounds__(128) scan_kernel(
    const float* __restrict__ input, const float* __restrict__ hx,
    const float* __restrict__ Wf,  const float* __restrict__ bf,
    const float* __restrict__ Wih, const float* __restrict__ bih,
    const float* __restrict__ bhh, const float* __restrict__ Wc,
    const float* __restrict__ bc,  const float* __restrict__ Wl,
    const float* __restrict__ bl,  const float* __restrict__ Wpi,
    const float* __restrict__ bpi, const float* __restrict__ Wb,
    const float* __restrict__ bb,  const float* __restrict__ S,
    const int* __restrict__ newep_flag, float* __restrict__ out) {
  const int n   = blockIdx.x;
  const int tid = threadIdx.x;

  __shared__ float sh_M[N_SUB][SUB_SIZE];
  __shared__ float sh_p[N_SUB], sh_r[SUB_SIZE], sh_h[HID], sh_g[SUB_SIZE];
  __shared__ float sh_s[HID], sh_hnew[HID];
  __shared__ float sh_l[3], sh_pc[N_SUB], sh_rn[SUB_SIZE];
  __shared__ float sh_lg[N_G];
  __shared__ float sh_b, sh_cg, sh_lpg, sh_lp;
  __shared__ int   sh_gidx;
  __shared__ float red[128];
  __shared__ int   redi[128];

  // Build M (constant over the scan; from t=0 task fields)
  if (tid < N_SUB * SUB_SIZE) {
    int i = tid / SUB_SIZE, k = tid % SUB_SIZE;
    const float* tp = input + (size_t)n * D_IN + CONV_OUT;
    int t1 = (int)tp[i];
    int c2 = (int)tp[N_SUB + i] - 1;
    int c3 = (int)tp[2 * N_SUB + i];
    float v;
    if (k < AA)           v = (k == t1) ? 1.0f : 0.0f;
    else if (k < AA + BB) v = (k - AA == c2) ? 1.0f : 0.0f;
    else                  v = (k - AA - BB == c3) ? 1.0f : 0.0f;
    sh_M[i][k] = v;
  }
  __syncthreads();

  const bool newep = (*newep_flag == 0);
  const float* hrow = hx + (size_t)n * STATE;
  if (tid < N_SUB)    sh_p[tid] = hrow[tid];
  if (tid < SUB_SIZE) sh_r[tid] = newep ? sh_M[0][tid] : hrow[5 + tid];
  if (tid < HID)      sh_h[tid] = hrow[19 + tid];
  if (tid < SUB_SIZE) sh_g[tid] = newep ? sh_M[0][tid] : hrow[83 + tid];
  if (tid == 0) {
    if (newep) sh_p[0] = 1.0f;
    sh_b = hrow[97];
  }
  __syncthreads();

  for (int t = 0; t < T_DIM; ++t) {
    const size_t srow = (size_t)(t * N_DIM + n) * S_W;
    // --- s = [x, r, g, b] @ Wf.T + bf  (x-part precomputed in S) ---
    if (tid < HID) {
      float acc = S[srow + tid] + bf[tid];
      const float* wf = Wf + (size_t)tid * WF_STRIDE + CONV_OUT;
#pragma unroll
      for (int k = 0; k < SUB_SIZE; ++k) acc += sh_r[k] * wf[k];
#pragma unroll
      for (int k = 0; k < SUB_SIZE; ++k) acc += sh_g[k] * wf[SUB_SIZE + k];
      acc += sh_b * wf[2 * SUB_SIZE];
      sh_s[tid] = acc;
    }
    __syncthreads();

    // --- cg = sigmoid([s,h] @ Wc.T + bc) ---
    float part = (tid < HID) ? (sh_s[tid] * Wc[tid] + sh_h[tid] * Wc[HID + tid]) : 0.0f;
    float dot = blk_sum(part, red, tid);
    if (tid == 0) sh_cg = sigm(dot + bc[0]);

    // --- GRU-ish gates -> hnew (uses old h) ---
    if (tid < HID) {
      const int j = tid;
      float gr = bih[j], gz = bih[HID + j], gn = bih[2 * HID + j];
      const float* w0 = Wih + (size_t)j * (2 * HID);
      const float* w1 = Wih + (size_t)(HID + j) * (2 * HID);
      const float* w2 = Wih + (size_t)(2 * HID + j) * (2 * HID);
      for (int k = 0; k < HID; ++k) {
        float sv = sh_s[k], hv = sh_h[k];
        gr += sv * w0[k] + hv * w0[HID + k];
        gz += sv * w1[k] + hv * w1[HID + k];
        gn += sv * w2[k] + hv * w2[HID + k];
      }
      float rg = sigm(gr + bhh[j]);
      float z  = sigm(gz + bhh[HID + j]);
      float nn = tanhf(gn + rg * bhh[2 * HID + j]);
      sh_hnew[j] = (1.0f - z) * nn;
    }
    // --- l = softmax(h_old @ Wl.T + bl) ---
    if (tid < 3) {
      float acc = bl[tid];
      const float* w = Wl + tid * HID;
      for (int k = 0; k < HID; ++k) acc += sh_h[k] * w[k];
      red[100 + tid] = acc;  // stash (red[0..] used later after sync)
    }
    __syncthreads();
    if (tid == 0) {
      float l0 = red[100], l1 = red[101], l2 = red[102];
      float m = fmaxf(l0, fmaxf(l1, l2));
      float e0 = expf(l0 - m), e1 = expf(l1 - m), e2 = expf(l2 - m);
      float inv = 1.0f / (e0 + e1 + e2);
      sh_l[0] = e0 * inv; sh_l[1] = e1 * inv; sh_l[2] = e2 * inv;
    }
    __syncthreads();

    // --- pconv ---
    if (tid < N_SUB) {
      float left  = (tid >= 1)         ? sh_p[tid - 1] : 0.0f;
      float right = (tid < N_SUB - 1)  ? sh_p[tid + 1] : 0.0f;
      sh_pc[tid] = left * sh_l[0] + sh_p[tid] * sh_l[1] + right * sh_l[2];
    }
    __syncthreads();
    // --- rnew = pconv @ M ---
    if (tid < SUB_SIZE) {
      float acc = 0.0f;
#pragma unroll
      for (int i = 0; i < N_SUB; ++i) acc += sh_pc[i] * sh_M[i][tid];
      sh_rn[tid] = acc;
    }
    __syncthreads();
    // --- commit p, r, h ---
    const float cg = sh_cg, ncg = 1.0f - sh_cg;
    if (tid < N_SUB)    sh_p[tid] = cg * sh_pc[tid] + ncg * sh_p[tid];
    if (tid < SUB_SIZE) sh_r[tid] = cg * sh_rn[tid] + ncg * sh_r[tid];
    if (tid < HID)      sh_h[tid] = cg * sh_hnew[tid] + ncg * sh_h[tid];
    __syncthreads();

    // --- lg = [h, r] @ Wpi.T + bpi ; sample via Gumbel-argmax ---
    float lgv = -INFINITY, lgn = -INFINITY;
    if (tid < N_G) {
      float acc = bpi[tid];
      const float* w = Wpi + (size_t)tid * (HID + SUB_SIZE);
      for (int k = 0; k < HID; ++k) acc += sh_h[k] * w[k];
#pragma unroll
      for (int k = 0; k < SUB_SIZE; ++k) acc += sh_r[k] * w[HID + k];
      sh_lg[tid] = acc;
      lgv = acc;
      lgn = acc + gumbel(t, n, tid, 0);
    }
    __syncthreads();
    int gidx = blk_argmax(lgn, red, redi, tid);
    float gmax = blk_max(lgv, red, tid);
    float se = blk_sum((tid < N_G) ? expf(sh_lg[tid] - gmax) : 0.0f, red, tid);
    if (tid == 0) {
      sh_gidx = gidx;
      sh_lpg = sh_lg[gidx] - (gmax + logf(se));
    }
    __syncthreads();

    // --- gemb, commit g ---
    if (tid < SUB_SIZE) {
      int gi = sh_gidx;
      int t1 = gi / (BB * CC);
      int rem = gi % (BB * CC);
      int c2 = rem / CC, c3 = rem % CC;
      float ge;
      if (tid < AA)           ge = (tid == t1) ? 1.0f : 0.0f;
      else if (tid < AA + BB) ge = (tid - AA == c2) ? 1.0f : 0.0f;
      else                    ge = (tid - AA - BB == c3) ? 1.0f : 0.0f;
      sh_g[tid] = cg * ge + ncg * sh_g[tid];
    }
    __syncthreads();

    // --- lbeta = [x, g] @ Wb.T + bb (x-part precomputed); sample b ---
    if (tid < 2) {
      float acc = S[srow + HID + tid] + bb[tid];
      const float* w = Wb + (size_t)tid * WB_STRIDE + CONV_OUT;
#pragma unroll
      for (int k = 0; k < SUB_SIZE; ++k) acc += sh_g[k] * w[k];
      red[110 + tid] = acc;
    }
    __syncthreads();
    if (tid == 0) {
      float l0 = red[110], l1 = red[111];
      float n0 = l0 + gumbel(t, n, 0, 1);
      float n1 = l1 + gumbel(t, n, 1, 1);
      int bi = (n1 > n0) ? 1 : 0;
      float m = fmaxf(l0, l1);
      float lse = m + logf(expf(l0 - m) + expf(l1 - m));
      sh_b = (float)bi;
      sh_lp = sh_lpg + ((bi ? l1 : l0) - lse);
    }
    __syncthreads();

    // --- emit output row ---
    if (tid < STATE) {
      float v;
      if (tid < 5)        v = sh_p[tid];
      else if (tid < 19)  v = sh_r[tid - 5];
      else if (tid < 83)  v = sh_h[tid - 19];
      else if (tid < 97)  v = sh_g[tid - 83];
      else if (tid == 97) v = sh_b;
      else                v = sh_lp;
      size_t base = ((size_t)t * N_DIM + n) * STATE;
      out[base + tid] = v;
      if (t == T_DIM - 1)
        out[(size_t)T_DIM * N_DIM * STATE + (size_t)n * STATE + tid] = v;
    }
    __syncthreads();
  }
}

// ---------------------------------------------------------------------------
extern "C" void kernel_launch(void* const* d_in, const int* in_sizes, int n_in,
                              void* d_out, int out_size, void* d_ws, size_t ws_size,
                              hipStream_t stream) {
  const float* input = (const float*)d_in[0];
  const float* hx    = (const float*)d_in[1];
  const float* Wf    = (const float*)d_in[2];
  const float* bf    = (const float*)d_in[3];
  const float* Wih   = (const float*)d_in[4];
  const float* bih   = (const float*)d_in[5];
  const float* bhh   = (const float*)d_in[6];
  const float* Wc    = (const float*)d_in[7];
  const float* bc    = (const float*)d_in[8];
  const float* Wl    = (const float*)d_in[9];
  const float* bl    = (const float*)d_in[10];
  const float* Wpi   = (const float*)d_in[11];
  const float* bpi   = (const float*)d_in[12];
  const float* Wb    = (const float*)d_in[13];
  const float* bb    = (const float*)d_in[14];
  float* out = (float*)d_out;

  char* ws = (char*)d_ws;
  _Float16* Wpack = (_Float16*)ws;                       // 655360 B
  int*      flag  = (int*)(ws + 0xA0000);                // 4 B (+pad)
  float*    S     = (float*)(ws + 0xA0000 + 256);        // 16384*80*4 = 5.24 MB

  hipMemsetAsync(flag, 0, sizeof(int), stream);
  pack_weights<<<(WPACK_ELEMS + 255) / 256, 256, 0, stream>>>(Wf, Wb, Wpack);
  check_hx<<<(N_DIM * STATE + 255) / 256, 256, 0, stream>>>(hx, flag);
  gemm_sx<<<128, 256, 0, stream>>>(input, Wpack, S);
  scan_kernel<<<N_DIM, 128, 0, stream>>>(input, hx, Wf, bf, Wih, bih, bhh,
                                         Wc, bc, Wl, bl, Wpi, bpi, Wb, bb,
                                         S, flag, out);
}
